// dissimpooling_71536975282899
// MI455X (gfx1250) — matrix operand order repaired and checked by reading, not compiled, not run on hardware
//
#include <hip/hip_runtime.h>
#include <hip/hip_bf16.h>
#include <stdint.h>

// ---------------- problem constants (match reference) ----------------
#define NN_   1024      // num nodes
#define EE_   16384     // num edges
#define DD_   256       // feature dim
#define MAXST 6
#define UB    2.0f

typedef __attribute__((ext_vector_type(16))) _Float16 v16h_t;
typedef __attribute__((ext_vector_type(8)))  _Float16 v8h_t;
typedef __attribute__((ext_vector_type(8)))  float    v8f_t;

// addrspace(3) pointer for async-to-LDS vdst operand (32-bit LDS address)
typedef __attribute__((address_space(3))) void* lds_ptr_t;

// ======================================================================
// Stage 0: min/max init
// ======================================================================
__global__ void k_init_mm(unsigned* mm) {
    if (threadIdx.x == 0) { mm[0] = 0x7f800000u; mm[1] = 0u; }
}

// ======================================================================
// Stage 1: per-edge L2 distance, one wave32 per edge; global min/max via
// monotone uint encoding (dist >= 0).
// ======================================================================
__global__ __launch_bounds__(256)
void k_edge_dist(const float* __restrict__ x, const long long* __restrict__ ei,
                 float* __restrict__ dist, unsigned* __restrict__ mm) {
    const int wave = (blockIdx.x << 3) + (threadIdx.x >> 5);   // 8 waves/block
    const int lane = threadIdx.x & 31;
    const int s = (int)ei[wave];
    const int d = (int)ei[EE_ + wave];
    const float* xs = x + (size_t)s * DD_;
    const float* xd = x + (size_t)d * DD_;
    float acc = 0.0f;
#pragma unroll
    for (int i = 0; i < DD_ / 32; i++) {
        float f = xs[lane + (i << 5)] - xd[lane + (i << 5)];
        acc = fmaf(f, f, acc);
    }
    for (int off = 16; off > 0; off >>= 1) acc += __shfl_xor(acc, off);
    if (lane == 0) {
        float r = sqrtf(acc);
        dist[wave] = r;
        atomicMin(&mm[0], __float_as_uint(r));
        atomicMax(&mm[1], __float_as_uint(r));
    }
}

// ======================================================================
// Stage 2: min-max normalize + 0.5, pack (src,dst) into u32
// ======================================================================
__global__ __launch_bounds__(256)
void k_norm_pack(const float* __restrict__ dist, const long long* __restrict__ ei,
                 const unsigned* __restrict__ mm,
                 float* __restrict__ score, unsigned* __restrict__ packed) {
    const int e = blockIdx.x * 256 + threadIdx.x;
    const float mn = __uint_as_float(mm[0]);
    const float mx = __uint_as_float(mm[1]);
    score[e]  = (dist[e] - mn) / (mx - mn) + 0.5f;
    packed[e] = ((unsigned)(int)ei[e]) | (((unsigned)(int)ei[EE_ + e]) << 16);
}

// ======================================================================
// Stage 3: greedy cluster search — one block (8 wave32s) per seed node.
// inc mask lives in LDS; argmin matches jnp.argmin (first min index).
// ======================================================================
__global__ __launch_bounds__(256)
void k_search(const float* __restrict__ score, const unsigned* __restrict__ packed,
              unsigned* __restrict__ bits,
              _Float16* __restrict__ aF16, _Float16* __restrict__ nsmF16,
              float* __restrict__ out_assign, float* __restrict__ out_nsm) {
    __shared__ unsigned inc[NN_ / 32];
    __shared__ float ws_s[8];
    __shared__ int   ws_e[8];
    __shared__ int   s_cont;
    const int tid  = threadIdx.x;
    const int seed = blockIdx.x;
    if (tid < NN_ / 32) inc[tid] = ((seed >> 5) == tid) ? (1u << (seed & 31)) : 0u;
    __syncthreads();

    float tot = 0.0f;
    int   ncnt = 0;
    int   nid[2 * MAXST];
    float nval[2 * MAXST];

    for (int step = 0; step < MAXST; step++) {
        float bs = 3.0e38f;
        int   be = 0x7fffffff;
        for (int e = tid; e < EE_; e += 256) {
            const unsigned p = packed[e];
            const int u = p & 0xFFFFu, v = p >> 16;
            const bool valid = ((inc[u >> 5] >> (u & 31)) & 1u) &&
                              !((inc[v >> 5] >> (v & 31)) & 1u);
            if (valid) {
                const float s = score[e];
                if (s < bs || (s == bs && e < be)) { bs = s; be = e; }
            }
        }
        for (int off = 16; off > 0; off >>= 1) {
            float os = __shfl_xor(bs, off);
            int   oe = __shfl_xor(be, off);
            if (os < bs || (os == bs && oe < be)) { bs = os; be = oe; }
        }
        if ((tid & 31) == 0) { ws_s[tid >> 5] = bs; ws_e[tid >> 5] = be; }
        __syncthreads();
        if (tid == 0) {
            float fbs = ws_s[0]; int fbe = ws_e[0];
            for (int w = 1; w < 8; w++)
                if (ws_s[w] < fbs || (ws_s[w] == fbs && ws_e[w] < fbe)) { fbs = ws_s[w]; fbe = ws_e[w]; }
            int active = (fbs < 1.0e30f) && (tot < UB);
            if (active) {
                const unsigned p = packed[fbe];
                const int u = p & 0xFFFFu, v = p >> 16;
                inc[v >> 5] |= 1u << (v & 31);
                const float hs = 0.5f * fbs;
                int fu = -1, fv = -1;
                for (int k = 0; k < ncnt; k++) { if (nid[k] == u) fu = k; if (nid[k] == v) fv = k; }
                if (fu >= 0) nval[fu] += hs; else { nid[ncnt] = u; nval[ncnt] = hs; ncnt++; }
                if (fv >= 0) nval[fv] += hs; else { nid[ncnt] = v; nval[ncnt] = hs; ncnt++; }
                tot += fbs;
            }
            s_cont = active;
        }
        __syncthreads();
        if (!s_cont) break;    // state frozen forever once inactive -> exact
    }
    __syncthreads();

    if (tid < NN_ / 32) bits[seed * (NN_ / 32) + tid] = inc[tid];
    const size_t rb = (size_t)seed * NN_;
    for (int c = tid; c < NN_; c += 256) {
        const float fb = (float)((inc[c >> 5] >> (c & 31)) & 1u);
        aF16[rb + c]      = (_Float16)fb;
        out_assign[rb + c] = fb;
        nsmF16[rb + c]    = (_Float16)0.0f;
        out_nsm[rb + c]   = 0.0f;
    }
    __threadfence_block();
    __syncthreads();
    if (tid == 0) {
        for (int k = 0; k < ncnt; k++) {
            nsmF16[rb + nid[k]]  = (_Float16)nval[k];
            out_nsm[rb + nid[k]] = nval[k];
        }
    }
}

// ======================================================================
// Stage 4: exact keep-first-unique-rows on 1024-bit masks
// ======================================================================
__global__ __launch_bounds__(128)
void k_dedup(const unsigned* __restrict__ bits, unsigned* __restrict__ keep) {
    __shared__ unsigned my[NN_ / 32];
    __shared__ int flag;
    const int i = blockIdx.x, tid = threadIdx.x;
    if (tid < NN_ / 32) my[tid] = bits[i * (NN_ / 32) + tid];
    if (tid == 0) flag = 0;
    __syncthreads();
    for (int j = tid; j < i; j += 128) {
        bool eq = true;
#pragma unroll
        for (int w = 0; w < NN_ / 32; w++) eq = eq && (bits[j * (NN_ / 32) + w] == my[w]);
        if (eq) flag = 1;
    }
    __syncthreads();
    if (tid == 0) keep[i] = flag ? 0u : 1u;
}

__global__ __launch_bounds__(256)
void k_mask(const unsigned* __restrict__ keep,
            _Float16* __restrict__ aF16, _Float16* __restrict__ nsmF16,
            float* __restrict__ out_assign, float* __restrict__ out_nsm) {
    const int row = blockIdx.x;
    if (keep[row]) return;
    const size_t rb = (size_t)row * NN_;
    for (int c = threadIdx.x; c < NN_; c += 256) {
        aF16[rb + c] = (_Float16)0.0f;  nsmF16[rb + c] = (_Float16)0.0f;
        out_assign[rb + c] = 0.0f;      out_nsm[rb + c] = 0.0f;
    }
}

// ======================================================================
// Stage 5: adjacency, stored TRANSPOSED (adjT[d][s]=1) so the GEMM's
// B-operand staging is contiguous.  Binarized adj is exact for
// (assign@adj@assignT)>0 since all terms are >= 0.  x is converted to
// f16 transposed (xT[n][k] = x[k][n]) for the same reason.
// ======================================================================
__global__ __launch_bounds__(256)
void k_zero_f16(_Float16* p, int n) {
    const int i = blockIdx.x * 256 + threadIdx.x;
    if (i < n) p[i] = (_Float16)0.0f;
}
__global__ __launch_bounds__(256)
void k_scatter_adjT(const unsigned* __restrict__ packed, _Float16* __restrict__ adjT) {
    const int e = blockIdx.x * 256 + threadIdx.x;
    const unsigned p = packed[e];
    adjT[(size_t)(p >> 16) * NN_ + (p & 0xFFFFu)] = (_Float16)1.0f;  // [dst][src]
}
__global__ __launch_bounds__(256)
void k_xT_f16(const float* __restrict__ x, _Float16* __restrict__ xT) {
    const int i = blockIdx.x * 256 + threadIdx.x;   // over N*D, coalesced read
    const int k = i >> 8;        // node index   (row of x)
    const int n = i & 255;       // feature index
    xT[(size_t)n * NN_ + k] = (_Float16)x[i];
}

// ======================================================================
// Stage 6: WMMA f16 GEMM, C(f32 acc) = A[M,K] @ Bt[Nn,K]^T.
// B is ALWAYS provided pre-transposed row-major [Nn,K] -> both LDS tiles
// stage as contiguous 16B chunks via CDNA5 global_load_async_to_lds_b128,
// double-buffered: tile k+1's async copies are in flight while tile k's
// WMMAs run; s_wait_asynccnt 4 retires exactly the current tile (ASYNCcnt
// completes in order, 4 ops/thread/tile).
//   EPIL 0: store f16   1: store f32   2: binarize(>0) + zero diag, f32
// Block: 128 thr = 4 waves; tile 64x64; wave 32x32 = 2x2 of v_wmma 16x16x32.
// Fragment layouts follow CDNA5 ISA 7.12.2 (16-bit A 16x32 / B 32x16).
// ======================================================================
template <int EPIL>
__global__ __launch_bounds__(128)
void k_gemm_wmma(const _Float16* __restrict__ A, int lda,
                 const _Float16* __restrict__ Bt, int ldb,
                 void* __restrict__ Cp, int ldc, int K) {
    __shared__ _Float16 As[2][64 * 40];   // [buf][row][k], pad 40 halfs (80B)
    __shared__ _Float16 Bs[2][64 * 40];   // [buf][n][k]
    const int tid  = threadIdx.x;
    const int bm   = blockIdx.y * 64;
    const int bn   = blockIdx.x * 64;
    const int wid  = tid >> 5, lane = tid & 31;
    const int wm   = (wid >> 1) * 32, wn = (wid & 1) * 32;
    const int sel  = lane >> 4;          // lane group 0/1
    const int ml   = lane & 15;          // A row / B col within 16-tile
    const int koffA = sel * 8;           // A halves: K[koff..+7], K[16+koff..+7]
    const int kbB   = sel * 8;           // B halves: K[kbB..+7], K[16+kbB..+7]

    // staging chunk owned by this thread (2 x 16B per matrix per tile)
    const int row0 = tid >> 2,          c40 = (tid & 3);
    const int row1 = (tid + 128) >> 2,  c41 = ((tid + 128) & 3);

    v8f_t acc[2][2] = {};

#define ASYNC_STAGE(kb, buf)                                                     \
    do {                                                                         \
        const _Float16* ga0 = A  + (size_t)(bm + row0) * lda + (kb) + c40 * 8;   \
        const _Float16* ga1 = A  + (size_t)(bm + row1) * lda + (kb) + c41 * 8;   \
        const _Float16* gb0 = Bt + (size_t)(bn + row0) * ldb + (kb) + c40 * 8;   \
        const _Float16* gb1 = Bt + (size_t)(bn + row1) * ldb + (kb) + c41 * 8;   \
        lds_ptr_t la0 = (lds_ptr_t)&As[buf][row0 * 40 + c40 * 8];                \
        lds_ptr_t la1 = (lds_ptr_t)&As[buf][row1 * 40 + c41 * 8];                \
        lds_ptr_t lb0 = (lds_ptr_t)&Bs[buf][row0 * 40 + c40 * 8];                \
        lds_ptr_t lb1 = (lds_ptr_t)&Bs[buf][row1 * 40 + c41 * 8];                \
        asm volatile("global_load_async_to_lds_b128 %0, %1, off"                 \
                     :: "v"(la0), "v"(ga0) : "memory");                          \
        asm volatile("global_load_async_to_lds_b128 %0, %1, off"                 \
                     :: "v"(la1), "v"(ga1) : "memory");                          \
        asm volatile("global_load_async_to_lds_b128 %0, %1, off"                 \
                     :: "v"(lb0), "v"(gb0) : "memory");                          \
        asm volatile("global_load_async_to_lds_b128 %0, %1, off"                 \
                     :: "v"(lb1), "v"(gb1) : "memory");                          \
    } while (0)

    ASYNC_STAGE(0, 0);                       // prologue: tile 0 in flight

    for (int kb = 0; kb < K; kb += 32) {
        const int buf = (kb >> 5) & 1;
        const bool more = (kb + 32) < K;
        if (more) ASYNC_STAGE(kb + 32, buf ^ 1);   // overlap next tile's copy
        if (more) asm volatile("s_wait_asynccnt 0x4" ::: "memory");
        else      asm volatile("s_wait_asynccnt 0x0" ::: "memory");
        __syncthreads();

        v16h_t af[2], bf[2];
#pragma unroll
        for (int i = 0; i < 2; i++) {
            const _Float16* ap = &As[buf][(wm + i * 16 + ml) * 40];
            v8h_t a0 = *reinterpret_cast<const v8h_t*>(ap + koffA);
            v8h_t a1 = *reinterpret_cast<const v8h_t*>(ap + 16 + koffA);
            af[i] = __builtin_shufflevector(a0, a1, 0,1,2,3,4,5,6,7,8,9,10,11,12,13,14,15);
        }
#pragma unroll
        for (int j = 0; j < 2; j++) {
            const _Float16* bp = &Bs[buf][(wn + j * 16 + ml) * 40 + kbB];
            v8h_t b0 = *reinterpret_cast<const v8h_t*>(bp);
            v8h_t b1 = *reinterpret_cast<const v8h_t*>(bp + 16);
            bf[j] = __builtin_shufflevector(b0, b1, 0,1,2,3,4,5,6,7,8,9,10,11,12,13,14,15);
        }
#pragma unroll
        for (int i = 0; i < 2; i++)
#pragma unroll
            for (int j = 0; j < 2; j++)
                acc[i][j] = __builtin_amdgcn_wmma_f32_16x16x32_f16(
                    false, af[i], false, bf[j], (short)0, acc[i][j], false, false);
        __syncthreads();   // all reads of buf done before it is restaged
    }
#undef ASYNC_STAGE

    // ---- epilogue: C/D layout = lane n=ml, row = sel*8 + v ----
#pragma unroll
    for (int i = 0; i < 2; i++)
#pragma unroll
        for (int j = 0; j < 2; j++)
#pragma unroll
            for (int v = 0; v < 8; v++) {
                const int gm = bm + wm + i * 16 + sel * 8 + v;
                const int gn = bn + wn + j * 16 + ml;
                const float val = acc[i][j][v];
                if (EPIL == 0) {
                    ((_Float16*)Cp)[(size_t)gm * ldc + gn] = (_Float16)val;
                } else if (EPIL == 1) {
                    ((float*)Cp)[(size_t)gm * ldc + gn] = val;
                } else {
                    float o = (val > 0.0f) ? 1.0f : 0.0f;
                    if (gm == gn) o = 0.0f;
                    ((float*)Cp)[(size_t)gm * ldc + gn] = o;
                }
            }
}

// ======================================================================
// host launcher
// ======================================================================
extern "C" void kernel_launch(void* const* d_in, const int* in_sizes, int n_in,
                              void* d_out, int out_size, void* d_ws, size_t ws_size,
                              hipStream_t stream) {
    (void)in_sizes; (void)n_in; (void)out_size; (void)ws_size;
    const float*     x  = (const float*)d_in[0];
    const long long* ei = (const long long*)d_in[1];

    // d_out: adj_new [N,N] | x_new [N,D] | assign_k [N,N] | nsm_k [N,N]
    float* out_adj    = (float*)d_out;
    float* out_x      = out_adj + (size_t)NN_ * NN_;
    float* out_assign = out_x + (size_t)NN_ * DD_;
    float* out_nsm    = out_assign + (size_t)NN_ * NN_;

    // workspace layout (all offsets 256B aligned), total ~8.8 MB
    char* W = (char*)d_ws;
    float*     dist   = (float*)(W + 0);            //  64 KB
    unsigned*  mm     = (unsigned*)(W + 65536);     //  256 B
    float*     score  = (float*)(W + 65792);        //  64 KB
    unsigned*  packed = (unsigned*)(W + 131328);    //  64 KB
    unsigned*  bits   = (unsigned*)(W + 196864);    // 128 KB
    unsigned*  keep   = (unsigned*)(W + 327936);    //   4 KB
    _Float16*  aF16   = (_Float16*)(W + 332032);    //   2 MB  assign
    _Float16*  nsmF16 = (_Float16*)(W + 2429184);   //   2 MB  nsm
    _Float16*  adjT16 = (_Float16*)(W + 4526336);   //   2 MB  adj^T (binary)
    _Float16*  tF16   = (_Float16*)(W + 6623488);   //   2 MB  assign@adj
    _Float16*  xT16   = (_Float16*)(W + 8720640);   // 512 KB  x^T

    k_init_mm<<<1, 32, 0, stream>>>(mm);
    k_edge_dist<<<EE_ / 8, 256, 0, stream>>>(x, ei, dist, mm);
    k_norm_pack<<<EE_ / 256, 256, 0, stream>>>(dist, ei, mm, score, packed);
    k_search<<<NN_, 256, 0, stream>>>(score, packed, bits, aF16, nsmF16,
                                      out_assign, out_nsm);
    k_dedup<<<NN_, 128, 0, stream>>>(bits, keep);
    k_mask<<<NN_, 256, 0, stream>>>(keep, aF16, nsmF16, out_assign, out_nsm);
    k_zero_f16<<<(NN_ * NN_) / 256, 256, 0, stream>>>(adjT16, NN_ * NN_);
    k_scatter_adjT<<<EE_ / 256, 256, 0, stream>>>(packed, adjT16);
    k_xT_f16<<<(NN_ * DD_) / 256, 256, 0, stream>>>(x, xT16);

    // T = assign_k @ adj      (Bt = adj^T; f16 out, exact small ints)
    k_gemm_wmma<0><<<dim3(NN_ / 64, NN_ / 64), 128, 0, stream>>>(
        aF16, NN_, adjT16, NN_, (void*)tF16, NN_, NN_);
    // adj_new = binarize(T @ assign_k^T) minus diagonal  (Bt = assign_k)
    k_gemm_wmma<2><<<dim3(NN_ / 64, NN_ / 64), 128, 0, stream>>>(
        tF16, NN_, aF16, NN_, (void*)out_adj, NN_, NN_);
    // x_new = nsm_k @ x       (Bt = x^T [D, N])
    k_gemm_wmma<1><<<dim3(DD_ / 64, NN_ / 64), 128, 0, stream>>>(
        nsmF16, NN_, xT16, NN_, (void*)out_x, DD_, NN_);
}
